// AggEncoder_40329742909862
// MI455X (gfx1250) — compile-verified, matrix operand order checked by reading, not run on hardware
//
#include <hip/hip_runtime.h>
#include <hip/hip_bf16.h>

// ---------------------------------------------------------------------------
// AggEncoder for MI455X (gfx1250, wave32, WMMA).
// Fused per-point MLP stack on v_wmma_f32_16x16x32_bf16.  PP=2 points share
// one workgroup so each B-fragment load from L2 feeds 2 WMMAs (halves L2
// weight traffic: ~19 GB -> ~9.6 GB for the 310-GFLOP dominant stage), while
// LDS stays at ~131 KB -> 2 workgroups / 320 KB WGP.
// ---------------------------------------------------------------------------

#define Bb 8
#define Nn 4096
#define Kk 16
#define Hh 64
#define PP 2           // points (16-row A blocks) per workgroup

typedef __attribute__((ext_vector_type(16))) __bf16 v16bf;
typedef __attribute__((ext_vector_type(8)))  float  v8f;

__device__ __forceinline__ __bf16 f2bf(float f) {
  unsigned u = __float_as_uint(f);
  unsigned r = (u + 0x7FFFu + ((u >> 16) & 1u)) >> 16;  // round-nearest-even
  unsigned short hs = (unsigned short)r;
  __bf16 o;
  __builtin_memcpy(&o, &hs, 2);
  return o;
}

// A-fragment (16xK bf16, ISA 7.12.2) position for element (row m, col k)
// within the per-kt 512-halfword block: lane = m + 16*g, halfword e.
__device__ __forceinline__ int afrag_index(int m, int col) {
  int kt = col >> 5;
  int kk = col & 31;
  int e    = (kk & 7) | ((kk >> 4) << 3);
  int lane = m + (((kk >> 3) & 1) << 4);
  return kt * 512 + lane * 16 + e;
}

// One wave computes N-tiles nt = wave, wave+8, ...  For each B fragment
// (loaded once from L2) it issues P WMMAs against P points' A fragments.
// Packed weights: tile (nt,kt), (lane,e) at ((nt*KT+kt)*32+lane)*16+e.
template <int P>
__device__ __forceinline__ void wave_gemm(const __bf16* __restrict__ aF,   // P blocks, stride 8192
                                          const __bf16* __restrict__ wP,
                                          int KT, int NT,
                                          float* accBuf, int W,            // P blocks, stride 8192
                                          const float* __restrict__ bias,
                                          int wave, int lane)
{
  for (int nt = wave; nt < NT; nt += 8) {
    v8f acc[P];
#pragma unroll
    for (int p = 0; p < P; ++p) { v8f z = {}; acc[p] = z; }
    const __bf16* bPtr = wP + (((nt * KT) * 32 + lane) * 16);
    const __bf16* aPtr = aF + lane * 16;
    for (int kt = 0; kt < KT; ++kt) {
      v16bf bv = *(const v16bf*)(bPtr + kt * 512);
#pragma unroll
      for (int p = 0; p < P; ++p) {
        v16bf av = *(const v16bf*)(aPtr + p * 8192 + kt * 512);
        acc[p] = __builtin_amdgcn_wmma_f32_16x16x32_bf16(
            false, av, false, bv, (short)0, acc[p], false, false);
      }
    }
    int col = nt * 16 + (lane & 15);
    float bb = bias[col];
    int mBase = (lane >> 4) * 8;  // C/D layout: VGPR r -> M = r + 8*(lane>>4)
#pragma unroll
    for (int p = 0; p < P; ++p)
#pragma unroll
      for (int r = 0; r < 8; ++r)
        accBuf[p * 8192 + (mBase + r) * W + col] = acc[p][r] + bb;
  }
}

// Row mean / inv-std for PP*16 = 32 rows in accBuf (8 threads per row).
__device__ __forceinline__ void row_stats(const float* buf, int W,
                                          float* redS, float* redQ,
                                          float* mrow, float* irow, int tid)
{
  int r = tid >> 3, t2 = tid & 7;          // r in [0,32)
  int p = r >> 4, rr = r & 15;
  const float* rowp = buf + p * 8192 + rr * W;
  float s = 0.f, q = 0.f;
  for (int c = t2; c < W; c += 8) { float v = rowp[c]; s += v; q += v * v; }
  redS[tid] = s; redQ[tid] = q;
  __syncthreads();
  if (t2 == 0) {
    float ss = 0.f, qq = 0.f;
    for (int i = 0; i < 8; ++i) { ss += redS[(r << 3) + i]; qq += redQ[(r << 3) + i]; }
    float m = ss / (float)W;
    float var = qq / (float)W - m * m;
    mrow[r] = m;
    irow[r] = rsqrtf(var + 1e-5f);
  }
  __syncthreads();
}

// ---------------------------------------------------------------------------
// Kernel: pack fp32 weight [Kd x Nd] into bf16 WMMA B-fragment order.
__global__ void pack_weight_kernel(const float* __restrict__ W,
                                   __bf16* __restrict__ dst, int Kd, int Nd)
{
  int i = blockIdx.x * blockDim.x + threadIdx.x;
  if (i >= Kd * Nd) return;
  int e    = i & 15;
  int lane = (i >> 4) & 31;
  int t    = i >> 9;
  int KT = Kd >> 5;
  int kt = t % KT, nt = t / KT;
  int Kr = kt * 32 + 16 * (e >> 3) + 8 * (lane >> 4) + (e & 7);
  int Nc = nt * 16 + (lane & 15);
  dst[i] = f2bf(W[Kr * Nd + Nc]);
}

// ---------------------------------------------------------------------------
// Kernel: embedding  pts = relu(LN(x @ W_emb + b_emb)); one block per point.
__global__ void embed_kernel(const float* __restrict__ x,
                             const float* __restrict__ W_emb,
                             const float* __restrict__ b_emb,
                             const float* __restrict__ g_emb,
                             const float* __restrict__ be_emb,
                             float* __restrict__ pts)
{
  int p = blockIdx.x;          // b*N + n
  int h = threadIdx.x;         // 64
  const float* xr = x + (long)p * 3;
  float v = xr[0] * W_emb[h] + xr[1] * W_emb[64 + h] + xr[2] * W_emb[128 + h] + b_emb[h];
  __shared__ float red[64];
  __shared__ float sm, sv;
  red[h] = v; __syncthreads();
  if (h == 0) { float s = 0.f; for (int i = 0; i < 64; ++i) s += red[i]; sm = s / 64.f; }
  __syncthreads();
  float d = v - sm;
  red[h] = d * d; __syncthreads();
  if (h == 0) { float s = 0.f; for (int i = 0; i < 64; ++i) s += red[i]; sv = s / 64.f; }
  __syncthreads();
  float o = d * rsqrtf(sv + 1e-5f) * g_emb[h] + be_emb[h];
  pts[(long)p * 64 + h] = fmaxf(o, 0.f);
}

// ---------------------------------------------------------------------------
// Kernel: brute-force KNN (top-16 smallest sq distance, stable by index).
__global__ void knn_kernel(const float* __restrict__ xyz, int* __restrict__ idx)
{
  int p = blockIdx.x * 128 + threadIdx.x;
  if (p >= Bb * Nn) return;
  int b = p >> 12;
  const float* base = xyz + (long)b * Nn * 3;
  float px = xyz[(long)p * 3 + 0], py = xyz[(long)p * 3 + 1], pz = xyz[(long)p * 3 + 2];
  __shared__ float dbuf[128][17];
  __shared__ int   ibuf[128][17];
  int t = threadIdx.x;
  for (int k = 0; k < 16; ++k) { dbuf[t][k] = 3.4e38f; ibuf[t][k] = 0; }
  for (int m = 0; m < Nn; ++m) {
    float dx = base[m * 3 + 0] - px;
    float dy = base[m * 3 + 1] - py;
    float dz = base[m * 3 + 2] - pz;
    float d = dx * dx + dy * dy + dz * dz;
    if (d < dbuf[t][15]) {
      int j = 15;
      while (j > 0 && dbuf[t][j - 1] > d) {
        dbuf[t][j] = dbuf[t][j - 1]; ibuf[t][j] = ibuf[t][j - 1]; --j;
      }
      dbuf[t][j] = d; ibuf[t][j] = m;
    }
  }
  for (int k = 0; k < 16; ++k) idx[(long)p * 16 + k] = ibuf[t][k];
}

// ---------------------------------------------------------------------------
// Kernel: per-batch sum / sumsq of diff = pts[idx] - pts (for global std).
__global__ void stats_kernel(const float* __restrict__ pts,
                             const int* __restrict__ idx,
                             float* __restrict__ stats)
{
  int b = blockIdx.y;
  int tid = threadIdx.x;
  const int total = Nn * Kk * Hh;  // 4194304
  float s = 0.f, q = 0.f;
  for (int i = blockIdx.x * 256 + tid; i < total; i += gridDim.x * 256) {
    int h = i & 63, k = (i >> 6) & 15, n = i >> 10;
    int nb = idx[((long)b * Nn + n) * 16 + k];
    float d = pts[((long)b * Nn + nb) * 64 + h] - pts[((long)b * Nn + n) * 64 + h];
    s += d; q += d * d;
  }
  __shared__ float rs[256], rq[256];
  rs[tid] = s; rq[tid] = q;
  __syncthreads();
  for (int off = 128; off > 0; off >>= 1) {
    if (tid < off) { rs[tid] += rs[tid + off]; rq[tid] += rq[tid + off]; }
    __syncthreads();
  }
  if (tid == 0) {
    atomicAdd(&stats[b * 2 + 0], rs[0]);
    atomicAdd(&stats[b * 2 + 1], rq[0]);
  }
}

__global__ void init_kernel(float* stats, float* y2)
{
  int i = blockIdx.x * blockDim.x + threadIdx.x;
  if (i < 16)   stats[i] = 0.f;
  if (i < 2048) y2[i]    = 0.f;
}

// ---------------------------------------------------------------------------
// Kernel A: 2 points/block: newp -> W_pre(+LN) -> resblock(W1a,W2a) -> max_K.
__global__ __launch_bounds__(256)
void point_block_kernel(const float* __restrict__ pts, const int* __restrict__ idx,
                        const float* __restrict__ stats,
                        const float* __restrict__ alpha, const float* __restrict__ beta,
                        const __bf16* __restrict__ Wpre, const float* __restrict__ bpre,
                        const float* __restrict__ gpre, const float* __restrict__ bepre,
                        const __bf16* __restrict__ W1,  const float* __restrict__ b1,
                        const float* __restrict__ g1,   const float* __restrict__ be1,
                        const __bf16* __restrict__ W2,  const float* __restrict__ b2,
                        const float* __restrict__ g2,   const float* __restrict__ be2,
                        float* __restrict__ y1)
{
  __shared__ __align__(32) __bf16 aFrag[PP * 8192];   // 32 KB
  __shared__ float accBuf[PP * 8192];                 // 64 KB  (16x512 / point)
  __shared__ float yRow[PP * 4096];                   // 32 KB  (16x256 / point)
  __shared__ float redS[256], redQ[256];
  __shared__ float mrow[PP * 16], irow[PP * 16];
  __shared__ float cRow[PP * 64];
  __shared__ int   nIdx[PP * 16];

  int tid  = threadIdx.x;
  int wave = tid >> 5, lane = tid & 31;
  int b  = blockIdx.x >> 11;             // 2048 blocks per batch
  int n0 = (blockIdx.x & 2047) * PP;

  // global per-batch std (ddof=1)
  float s = stats[b * 2 + 0], q = stats[b * 2 + 1];
  const float cnt = (float)(Nn * Kk * Hh);
  float var = (q - s * s / cnt) / (cnt - 1.f);
  float invStd = 1.f / (sqrtf(fmaxf(var, 0.f)) + 1e-5f);

  if (tid < PP * 64) {
    int p = tid >> 6, h = tid & 63;
    cRow[tid] = pts[((long)b * Nn + n0 + p) * 64 + h];
  }
  if (tid < PP * 16) {
    int p = tid >> 4, k = tid & 15;
    nIdx[tid] = idx[((long)b * Nn + n0 + p) * 16 + k];
  }
  __syncthreads();

  // build newp [16 x 128] per point, directly in A-fragment layout
  for (int i = tid; i < PP * 16 * 128; i += 256) {
    int p = i >> 11, rem = i & 2047;
    int m = rem >> 7, col = rem & 127;
    float v;
    if (col < 64) {
      float nbv = pts[((long)b * Nn + nIdx[p * 16 + m]) * 64 + col];
      v = (nbv - cRow[p * 64 + col]) * invStd * alpha[col] + beta[col];
    } else {
      v = cRow[p * 64 + col - 64];
    }
    aFrag[p * 8192 + afrag_index(m, col)] = f2bf(v);
  }
  __syncthreads();

  // GEMM1: [16x128]x[128x256] + bias, then LN
  wave_gemm<PP>(aFrag, Wpre, 4, 16, accBuf, 256, bpre, wave, lane);
  __syncthreads();
  row_stats(accBuf, 256, redS, redQ, mrow, irow, tid);
  {
    int r = tid >> 3, t2 = tid & 7;
    int p = r >> 4, rr = r & 15;
    for (int c = t2; c < 256; c += 8) {
      float v = (accBuf[p * 8192 + rr * 256 + c] - mrow[r]) * irow[r] * gpre[c] + bepre[c];
      yRow[p * 4096 + rr * 256 + c] = v;                  // resblock residual
      aFrag[p * 8192 + afrag_index(rr, c)] = f2bf(v);     // A for GEMM2
    }
  }
  __syncthreads();

  // GEMM2: [16x256]x[256x512] + bias, LN, relu
  wave_gemm<PP>(aFrag, W1, 8, 32, accBuf, 512, b1, wave, lane);
  __syncthreads();
  row_stats(accBuf, 512, redS, redQ, mrow, irow, tid);
  {
    int r = tid >> 3, t2 = tid & 7;
    int p = r >> 4, rr = r & 15;
    for (int c = t2; c < 512; c += 8) {
      float v = (accBuf[p * 8192 + rr * 512 + c] - mrow[r]) * irow[r] * g1[c] + be1[c];
      aFrag[p * 8192 + afrag_index(rr, c)] = f2bf(fmaxf(v, 0.f));
    }
  }
  __syncthreads();

  // GEMM3: [16x512]x[512x256] + bias, LN, +residual, relu
  wave_gemm<PP>(aFrag, W2, 16, 16, accBuf, 256, b2, wave, lane);
  __syncthreads();
  row_stats(accBuf, 256, redS, redQ, mrow, irow, tid);
  {
    int r = tid >> 3, t2 = tid & 7;
    int p = r >> 4, rr = r & 15;
    for (int c = t2; c < 256; c += 8) {
      float v = (accBuf[p * 8192 + rr * 256 + c] - mrow[r]) * irow[r] * g2[c] + be2[c];
      accBuf[p * 8192 + rr * 256 + c] = fmaxf(yRow[p * 4096 + rr * 256 + c] + v, 0.f);
    }
  }
  __syncthreads();

  // max-pool over the 16 neighbor rows, per point
  for (int i = tid; i < PP * 256; i += 256) {
    int p = i >> 8, col = i & 255;
    float mx = accBuf[p * 8192 + col];
    for (int r = 1; r < 16; ++r) mx = fmaxf(mx, accBuf[p * 8192 + r * 256 + col]);
    y1[((long)b * Nn + n0 + p) * 256 + col] = mx;
  }
}

// ---------------------------------------------------------------------------
// Kernel B: 32 points/block (2 row-blocks): W_pos(+LN) -> resblock -> max_N.
__global__ __launch_bounds__(256)
void pos_block_kernel(const float* __restrict__ y1,
                      const __bf16* __restrict__ Wpos, const float* __restrict__ bpos,
                      const float* __restrict__ gpos,  const float* __restrict__ bepos,
                      const __bf16* __restrict__ W1,   const float* __restrict__ b1,
                      const float* __restrict__ g1,    const float* __restrict__ be1,
                      const __bf16* __restrict__ W2,   const float* __restrict__ b2,
                      const float* __restrict__ g2,    const float* __restrict__ be2,
                      float* __restrict__ y2)
{
  __shared__ __align__(32) __bf16 aFrag[PP * 8192];
  __shared__ float accBuf[PP * 8192];
  __shared__ float yRow[PP * 4096];
  __shared__ float redS[256], redQ[256];
  __shared__ float mrow[PP * 16], irow[PP * 16];

  int tid  = threadIdx.x;
  int wave = tid >> 5, lane = tid & 31;
  int b  = blockIdx.x >> 7;              // 128 blocks per batch
  int n0 = (blockIdx.x & 127) * (PP * 16);

  for (int i = tid; i < PP * 16 * 256; i += 256) {
    int row = i >> 8, col = i & 255;
    int p = row >> 4, m = row & 15;
    aFrag[p * 8192 + afrag_index(m, col)] =
        f2bf(y1[((long)b * Nn + n0 + row) * 256 + col]);
  }
  __syncthreads();

  wave_gemm<PP>(aFrag, Wpos, 8, 16, accBuf, 256, bpos, wave, lane);
  __syncthreads();
  row_stats(accBuf, 256, redS, redQ, mrow, irow, tid);
  {
    int r = tid >> 3, t2 = tid & 7;
    int p = r >> 4, rr = r & 15;
    for (int c = t2; c < 256; c += 8) {
      float v = (accBuf[p * 8192 + rr * 256 + c] - mrow[r]) * irow[r] * gpos[c] + bepos[c];
      yRow[p * 4096 + rr * 256 + c] = v;
      aFrag[p * 8192 + afrag_index(rr, c)] = f2bf(v);
    }
  }
  __syncthreads();

  wave_gemm<PP>(aFrag, W1, 8, 32, accBuf, 512, b1, wave, lane);
  __syncthreads();
  row_stats(accBuf, 512, redS, redQ, mrow, irow, tid);
  {
    int r = tid >> 3, t2 = tid & 7;
    int p = r >> 4, rr = r & 15;
    for (int c = t2; c < 512; c += 8) {
      float v = (accBuf[p * 8192 + rr * 512 + c] - mrow[r]) * irow[r] * g1[c] + be1[c];
      aFrag[p * 8192 + afrag_index(rr, c)] = f2bf(fmaxf(v, 0.f));
    }
  }
  __syncthreads();

  wave_gemm<PP>(aFrag, W2, 16, 16, accBuf, 256, b2, wave, lane);
  __syncthreads();
  row_stats(accBuf, 256, redS, redQ, mrow, irow, tid);
  {
    int r = tid >> 3, t2 = tid & 7;
    int p = r >> 4, rr = r & 15;
    for (int c = t2; c < 256; c += 8) {
      float v = (accBuf[p * 8192 + rr * 256 + c] - mrow[r]) * irow[r] * g2[c] + be2[c];
      accBuf[p * 8192 + rr * 256 + c] = fmaxf(yRow[p * 4096 + rr * 256 + c] + v, 0.f);
    }
  }
  __syncthreads();

  // partial max over rows; values >= 0 so uint atomicMax is order-preserving
  for (int i = tid; i < PP * 256; i += 256) {
    int p = i >> 8, col = i & 255;
    float mx = accBuf[p * 8192 + col];
    for (int r = 1; r < 16; ++r) mx = fmaxf(mx, accBuf[p * 8192 + r * 256 + col]);
    atomicMax((unsigned int*)&y2[b * 256 + col], __float_as_uint(mx));
  }
}

// ---------------------------------------------------------------------------
// Kernel C: final  out = LN(y2 @ W_fin + b_fin); tiny (8x256x64), plain VALU.
__global__ __launch_bounds__(512)
void final_kernel(const float* __restrict__ y2, const float* __restrict__ W_fin,
                  const float* __restrict__ b_fin, const float* __restrict__ g_fin,
                  const float* __restrict__ be_fin, float* __restrict__ out)
{
  int tid = threadIdx.x;          // 512 = 8 rows x 64 cols
  int row = tid >> 6, col = tid & 63;
  float acc = b_fin[col];
  for (int k = 0; k < 256; ++k) acc += y2[row * 256 + k] * W_fin[k * 64 + col];
  __shared__ float buf[8][64];
  __shared__ float mean[8], inv[8];
  buf[row][col] = acc;
  __syncthreads();
  if (col == 0) {
    float s = 0.f;
    for (int i = 0; i < 64; ++i) s += buf[row][i];
    float m = s / 64.f, qq = 0.f;
    for (int i = 0; i < 64; ++i) { float d = buf[row][i] - m; qq += d * d; }
    mean[row] = m;
    inv[row]  = rsqrtf(qq / 64.f + 1e-5f);
  }
  __syncthreads();
  out[row * 64 + col] = (acc - mean[row]) * inv[row] * g_fin[col] + be_fin[col];
}

// ---------------------------------------------------------------------------
extern "C" void kernel_launch(void* const* d_in, const int* in_sizes, int n_in,
                              void* d_out, int out_size, void* d_ws, size_t ws_size,
                              hipStream_t stream)
{
  (void)in_sizes; (void)n_in; (void)out_size; (void)ws_size;
  const float* xyz    = (const float*)d_in[0];
  const float* x      = (const float*)d_in[1];
  const float* W_emb  = (const float*)d_in[2];
  const float* b_emb  = (const float*)d_in[3];
  const float* g_emb  = (const float*)d_in[4];
  const float* be_emb = (const float*)d_in[5];
  const float* alpha  = (const float*)d_in[6];
  const float* beta   = (const float*)d_in[7];
  const float* W_pre  = (const float*)d_in[8];
  const float* b_pre  = (const float*)d_in[9];
  const float* g_pre  = (const float*)d_in[10];
  const float* be_pre = (const float*)d_in[11];
  const float* W1a    = (const float*)d_in[12];
  const float* b1a    = (const float*)d_in[13];
  const float* g1a    = (const float*)d_in[14];
  const float* be1a   = (const float*)d_in[15];
  const float* W2a    = (const float*)d_in[16];
  const float* b2a    = (const float*)d_in[17];
  const float* g2a    = (const float*)d_in[18];
  const float* be2a   = (const float*)d_in[19];
  const float* W_pos  = (const float*)d_in[20];
  const float* b_pos  = (const float*)d_in[21];
  const float* g_pos  = (const float*)d_in[22];
  const float* be_pos = (const float*)d_in[23];
  const float* W1b    = (const float*)d_in[24];
  const float* b1b    = (const float*)d_in[25];
  const float* g1b    = (const float*)d_in[26];
  const float* be1b   = (const float*)d_in[27];
  const float* W2b    = (const float*)d_in[28];
  const float* b2b    = (const float*)d_in[29];
  const float* g2b    = (const float*)d_in[30];
  const float* be2b   = (const float*)d_in[31];
  const float* W_fin  = (const float*)d_in[32];
  const float* b_fin  = (const float*)d_in[33];
  const float* g_fin  = (const float*)d_in[34];
  const float* be_fin = (const float*)d_in[35];

  char* ws = (char*)d_ws;
  size_t off = 0;
  float* pts   = (float*)(ws + off); off += (size_t)Bb * Nn * 64 * 4;   // 8 MB
  int*   idx   = (int*)  (ws + off); off += (size_t)Bb * Nn * 16 * 4;   // 2 MB
  float* stats = (float*)(ws + off); off += 256;
  float* y2    = (float*)(ws + off); off += (size_t)Bb * 256 * 4;       // 8 KB
  float* y1    = (float*)(ws + off); off += (size_t)Bb * Nn * 256 * 4;  // 32 MB
  __bf16* pWpre = (__bf16*)(ws + off); off += (size_t)128 * 256 * 2;
  __bf16* pW1a  = (__bf16*)(ws + off); off += (size_t)256 * 512 * 2;
  __bf16* pW2a  = (__bf16*)(ws + off); off += (size_t)512 * 256 * 2;
  __bf16* pWpos = (__bf16*)(ws + off); off += (size_t)256 * 256 * 2;
  __bf16* pW1b  = (__bf16*)(ws + off); off += (size_t)256 * 512 * 2;
  __bf16* pW2b  = (__bf16*)(ws + off); off += (size_t)512 * 256 * 2;

  init_kernel<<<8, 256, 0, stream>>>(stats, y2);

  pack_weight_kernel<<<(128 * 256) / 256, 256, 0, stream>>>(W_pre, pWpre, 128, 256);
  pack_weight_kernel<<<(256 * 512) / 256, 256, 0, stream>>>(W1a,   pW1a,  256, 512);
  pack_weight_kernel<<<(512 * 256) / 256, 256, 0, stream>>>(W2a,   pW2a,  512, 256);
  pack_weight_kernel<<<(256 * 256) / 256, 256, 0, stream>>>(W_pos, pWpos, 256, 256);
  pack_weight_kernel<<<(256 * 512) / 256, 256, 0, stream>>>(W1b,   pW1b,  256, 512);
  pack_weight_kernel<<<(512 * 256) / 256, 256, 0, stream>>>(W2b,   pW2b,  512, 256);

  embed_kernel<<<Bb * Nn, 64, 0, stream>>>(x, W_emb, b_emb, g_emb, be_emb, pts);
  knn_kernel<<<(Bb * Nn) / 128, 128, 0, stream>>>(xyz, idx);
  {
    dim3 g(256, Bb);
    stats_kernel<<<g, 256, 0, stream>>>(pts, idx, stats);
  }

  point_block_kernel<<<(Bb * Nn) / PP, 256, 0, stream>>>(
      pts, idx, stats, alpha, beta,
      pWpre, b_pre, g_pre, be_pre,
      pW1a, b1a, g1a, be1a,
      pW2a, b2a, g2a, be2a, y1);

  pos_block_kernel<<<(Bb * Nn) / (PP * 16), 256, 0, stream>>>(
      y1,
      pWpos, b_pos, g_pos, be_pos,
      pW1b, b1b, g1b, be1b,
      pW2b, b2b, g2b, be2b, y2);

  final_kernel<<<1, 512, 0, stream>>>(y2, W_fin, b_fin, g_fin, be_fin, (float*)d_out);
}
